// LidarEncoder_17248588661259
// MI455X (gfx1250) — matrix operand order repaired and checked by reading, not compiled, not run on hardware
//
#include <hip/hip_runtime.h>
#include <hip/hip_bf16.h>

// ---------------------------------------------------------------------------
// LidarEncoder on gfx1250: voxelize -> 11x (conv3d + masked BN + ReLU) -> pool
// Conv layers 1..10 are implicit-GEMM over v_wmma_f32_16x16x32_f16.
// Each wave holds NPW cout tiles (NPW accumulators) and reuses the A fragment
// across them, amortizing the scattered activation loads + tap address math.
// ---------------------------------------------------------------------------

typedef __attribute__((ext_vector_type(16))) _Float16 v16h;
typedef __attribute__((ext_vector_type(8)))  _Float16 v8h;
typedef __attribute__((ext_vector_type(8)))  float    v8f;

#define BATCH 2
#define G64   64
#define GRID3 (G64*G64*G64)

static __device__ __forceinline__ v16h join8(v8h lo, v8h hi) {
  return __builtin_shufflevector(lo, hi, 0,1,2,3,4,5,6,7,8,9,10,11,12,13,14,15);
}

// ------------------------------- voxelize ----------------------------------

__global__ void zero_kernel(float* __restrict__ p, int n) {
  int i = blockIdx.x * blockDim.x + threadIdx.x;
  if (i < n) p[i] = 0.0f;
}

__global__ void vox_scatter_kernel(const float* __restrict__ pts,
                                   float* __restrict__ sums,
                                   float* __restrict__ cnts, int Npts) {
  int i = blockIdx.x * blockDim.x + threadIdx.x;
  if (i >= BATCH * Npts) return;
  const float* p = pts + (long long)i * 4;
  float x = p[0], y = p[1], z = p[2], inten = p[3];
  float valid = (x != 0.0f && y != 0.0f && z != 0.0f) ? 1.0f : 0.0f;
  int b = i / Npts;
  int cx = (int)rintf(x * 20.0f) + 32; cx = cx < 0 ? 0 : (cx > 63 ? 63 : cx);
  int cy = (int)rintf(y * 20.0f) + 32; cy = cy < 0 ? 0 : (cy > 63 ? 63 : cy);
  int cz = (int)rintf(z * 20.0f) + 32; cz = cz < 0 ? 0 : (cz > 63 ? 63 : cz);
  long long flat = (((long long)b * 64 + cx) * 64 + cy) * 64 + cz;
  atomicAdd(&sums[flat], inten * valid);
  atomicAdd(&cnts[flat], valid);
}

__global__ void vox_finalize_kernel(const float* __restrict__ sums,
                                    const float* __restrict__ cnts,
                                    float* __restrict__ feat,
                                    float* __restrict__ occ) {
  int i = blockIdx.x * blockDim.x + threadIdx.x;
  if (i >= BATCH * GRID3) return;
  float c = cnts[i];
  feat[i] = sums[i] / fmaxf(c, 1.0f);
  occ[i]  = (c > 0.0f) ? 1.0f : 0.0f;
}

// ---------------------------- weight packing -------------------------------
// Pack fp32 OIDHW weights into f16 blocks of 512 elements laid out exactly in
// this code's contiguous-K WMMA B-matrix (K=32 x N=16) lane order (matches the A loads):
//   block = ((t*kChunks + kc)*nTiles + nt)
//   lane l (0..31): column N = nt*16 + (l&15); element j: K = (l<16 ? j : 16+j)

__global__ void pack_weights_kernel(const float* __restrict__ w,
                                    _Float16* __restrict__ dst,
                                    int Cin, int Cout, int taps) {
  long long e = (long long)blockIdx.x * blockDim.x + threadIdx.x;
  long long total = (long long)taps * Cin * Cout;
  if (e >= total) return;
  int j    = (int)(e & 15);
  int lane = (int)((e >> 4) & 31);
  long long blk = e >> 9;
  int nTiles = Cout >> 4, kChunks = Cin >> 5;
  int nt = (int)(blk % nTiles);
  long long tmp = blk / nTiles;
  int kc = (int)(tmp % kChunks);
  int t  = (int)(tmp / kChunks);
  int n = nt * 16 + (lane & 15);
  int K = (lane < 16) ? j : (16 + j);
  int c = kc * 32 + K;
  long long src = ((long long)n * Cin + c) * taps + t;
  dst[e] = (_Float16)w[src];
}

// -------------------------- layer 0: direct conv ---------------------------

__global__ void conv0_kernel(const float* __restrict__ feat,
                             const float* __restrict__ w,   // [32,1,3,3,3]
                             float* __restrict__ out) {     // [B*64^3, 32]
  int p = blockIdx.x * blockDim.x + threadIdx.x;
  if (p >= BATCH * GRID3) return;
  int b = p >> 18;
  int r = p & (GRID3 - 1);
  int z = r >> 12, y = (r >> 6) & 63, x = r & 63;
  float v[27];
#pragma unroll
  for (int t = 0; t < 27; ++t) {
    int dz = t / 9 - 1, rr = t % 9;
    int dy = rr / 3 - 1, dx = rr % 3 - 1;
    int iz = z + dz, iy = y + dy, ix = x + dx;
    bool inb = ((unsigned)iz < 64u) && ((unsigned)iy < 64u) && ((unsigned)ix < 64u);
    v[t] = inb ? feat[(((b * 64 + iz) * 64 + iy) * 64) + ix] : 0.0f;
  }
  for (int co = 0; co < 32; ++co) {
    float s = 0.0f;
#pragma unroll
    for (int t = 0; t < 27; ++t) s += w[co * 27 + t] * v[t];
    out[(long long)p * 32 + co] = s;
  }
}

// ------------------------ implicit-GEMM WMMA conv --------------------------
// act: f16 channels-last [B, Si, Si, Si, Cin]; out: f32 [B, So, So, So, Cout]
// One wave per 16(M spatial) x (NPW*16)(N cout) output tile.
// K loop: taps x Cin/32. A fragment loaded once per K-step, reused NPW times.

template <int NPW>
__global__ __launch_bounds__(32)
void conv_wmma_kernel(const _Float16* __restrict__ act,
                      const _Float16* __restrict__ wpk,
                      float* __restrict__ out,
                      int logSi, int logSo, int stride, int taps,
                      int Cin, int Cout) {
  const int lane = threadIdx.x;
  const int half = lane >> 4;
  const int m    = lane & 15;
  const int Si = 1 << logSi, So = 1 << logSo;
  const int p  = blockIdx.x * 16 + m;                 // output voxel (A row M)
  const int b  = p >> (3 * logSo);
  int r        = p & ((1 << (3 * logSo)) - 1);
  const int z  = r >> (2 * logSo);
  const int y  = (r >> logSo) & (So - 1);
  const int x  = r & (So - 1);
  const int kChunks = Cin >> 5;
  const int nTiles  = Cout >> 4;
  const int nt0     = blockIdx.y * NPW;
  const int runA0 = half ? 16 : 0;    // elements 0..7  -> K base
  const int runA1 = half ? 24 : 8;    // elements 8..15 -> K base

  v8f acc[NPW];
#pragma unroll
  for (int q = 0; q < NPW; ++q) acc[q] = (v8f){};
  v8h zero8 = {};

  for (int t = 0; t < taps; ++t) {
    int dz = 0, dy = 0, dx = 0;
    if (taps == 27) {
      dz = t / 9 - 1;
      int rr = t % 9;
      dy = rr / 3 - 1;
      dx = rr % 3 - 1;
    }
    const int iz = z * stride + dz, iy = y * stride + dy, ix = x * stride + dx;
    const bool inb = ((unsigned)iz < (unsigned)Si) && ((unsigned)iy < (unsigned)Si) &&
                     ((unsigned)ix < (unsigned)Si);
    const long long vbase =
        (((long long)(b * Si + iz) * Si + iy) * Si + ix) * (long long)Cin;
    const _Float16* wbase = wpk + ((long long)t * kChunks * nTiles) * 512;
    for (int kc = 0; kc < kChunks; ++kc) {
      v8h a0 = zero8, a1 = zero8;
      if (inb) {
        const _Float16* ap = act + vbase + kc * 32;
        a0 = *(const v8h*)(ap + runA0);
        a1 = *(const v8h*)(ap + runA1);
      }
      v16h av = join8(a0, a1);
      const _Float16* wrow =
          wbase + ((long long)(kc * nTiles + nt0)) * 512 + lane * 16;
#pragma unroll
      for (int q = 0; q < NPW; ++q) {
        const _Float16* wp = wrow + (long long)q * 512;
        v16h bv = join8(*(const v8h*)wp, *(const v8h*)(wp + 8));
        acc[q] = __builtin_amdgcn_wmma_f32_16x16x32_f16(
            false, av, false, bv, (short)0, acc[q], false, false);
      }
    }
  }
  // D layout: lane l -> N = l&15; VGPR r -> M = r + 8*(l>=16). p' is linear.
  const long long pbase = (long long)blockIdx.x * 16 + half * 8;
#pragma unroll
  for (int q = 0; q < NPW; ++q) {
    const int n = (nt0 + q) * 16 + (lane & 15);
#pragma unroll
    for (int rr = 0; rr < 8; ++rr)
      out[(pbase + rr) * Cout + n] = acc[q][rr];
  }
}

// ----------------------- masked BN: stats + apply --------------------------

__global__ void bn_stats_kernel(const float* __restrict__ x,
                                const float* __restrict__ occ,
                                float* __restrict__ stats,  // [C]=sum [C]=sumsq [1]=cnt
                                int nvox, int C, int logC) {
  __shared__ float ssum[512];
  __shared__ float ssq[512];
  __shared__ float scnt;
  for (int i = threadIdx.x; i < C; i += blockDim.x) { ssum[i] = 0.f; ssq[i] = 0.f; }
  if (threadIdx.x == 0) scnt = 0.f;
  __syncthreads();
  long long total  = (long long)nvox * C;
  long long stride = (long long)gridDim.x * blockDim.x;
  for (long long e = (long long)blockIdx.x * blockDim.x + threadIdx.x; e < total;
       e += stride) {
    int c = (int)(e & (C - 1));
    long long vox = e >> logC;
    float o  = occ[vox];
    float xv = x[e];
    float xm = xv * o;
    atomicAdd(&ssum[c], xm);
    atomicAdd(&ssq[c], xm * xv);
    if (c == 0) atomicAdd(&scnt, o);
  }
  __syncthreads();
  for (int i = threadIdx.x; i < C; i += blockDim.x) {
    atomicAdd(&stats[i], ssum[i]);
    atomicAdd(&stats[C + i], ssq[i]);
  }
  if (threadIdx.x == 0) atomicAdd(&stats[2 * C], scnt);
}

__global__ void bn_apply_kernel(const float* __restrict__ x,
                                const float* __restrict__ occ,
                                const float* __restrict__ stats,
                                const float* __restrict__ gamma,
                                const float* __restrict__ beta,
                                _Float16* __restrict__ y,
                                int nvox, int C, int logC) {
  long long total = (long long)nvox * C;
  long long e = (long long)blockIdx.x * blockDim.x + threadIdx.x;
  if (e >= total) return;
  int c = (int)(e & (C - 1));
  long long vox = e >> logC;
  float cnt  = fmaxf(stats[2 * C], 1.0f);
  float mean = stats[c] / cnt;
  float var  = fmaxf(stats[C + c] / cnt - mean * mean, 0.0f);
  float rs   = rsqrtf(var + 1e-5f);
  float val  = fmaxf(gamma[c] * ((x[e] - mean) * rs) + beta[c], 0.0f) * occ[vox];
  y[e] = (_Float16)val;
}

// ------------------------- occupancy downsample ----------------------------

__global__ void occ_down_kernel(const float* __restrict__ oin,
                                float* __restrict__ oout, int logSo) {
  int So = 1 << logSo;
  int p = blockIdx.x * blockDim.x + threadIdx.x;
  int n = BATCH << (3 * logSo);
  if (p >= n) return;
  int b = p >> (3 * logSo);
  int r = p & ((1 << (3 * logSo)) - 1);
  int z = r >> (2 * logSo);
  int y = (r >> logSo) & (So - 1);
  int x = r & (So - 1);
  int Si = So * 2;
  float m = 0.0f;
#pragma unroll
  for (int d = 0; d < 8; ++d) {
    int dz = d >> 2, dy = (d >> 1) & 1, dx = d & 1;
    m = fmaxf(m, oin[(((b * Si + 2 * z + dz) * Si + 2 * y + dy) * Si + 2 * x + dx)]);
  }
  oout[p] = m;
}

// ------------------------------ final pool ---------------------------------

__global__ void pool_kernel(const _Float16* __restrict__ act,  // [B*64, 512]
                            const float* __restrict__ occ4,    // [B*64]
                            float* __restrict__ out) {         // [B, 512]
  int b = blockIdx.x;
  int c = threadIdx.x;
  float s = 0.0f, oc = 0.0f;
  for (int v = 0; v < 64; ++v) {
    s  += (float)act[((long long)(b * 64 + v)) * 512 + c];
    oc += occ4[b * 64 + v];
  }
  out[b * 512 + c] = s / fmaxf(oc, 1.0f);
}

// ------------------------------- launcher ----------------------------------

extern "C" void kernel_launch(void* const* d_in, const int* in_sizes, int n_in,
                              void* d_out, int out_size, void* d_ws, size_t ws_size,
                              hipStream_t stream) {
  const float* points = (const float*)d_in[0];
  const float* W[11];
  const float* Gm[11];
  const float* Bt[11];
  for (int i = 0; i < 11; ++i) {
    W[i]  = (const float*)d_in[1 + i];
    Gm[i] = (const float*)d_in[12 + i];
    Bt[i] = (const float*)d_in[23 + i];
  }
  const int Npts = in_sizes[0] / (BATCH * 4);

  // ------- layer table -------
  const int cins[11]  = {1, 32, 32, 64, 64, 128, 128, 256, 256, 512, 512};
  const int couts[11] = {32, 32, 64, 64, 128, 128, 256, 256, 512, 512, 512};
  const int ks[11]    = {3, 3, 3, 3, 3, 3, 3, 3, 3, 3, 1};
  const int ss[11]    = {1, 1, 2, 1, 2, 1, 2, 1, 2, 1, 1};
  const int Sin[11]   = {64, 64, 64, 32, 32, 16, 16, 8, 8, 4, 4};
  const int Sout[11]  = {64, 64, 32, 32, 16, 16, 8, 8, 4, 4, 4};

  // ------- workspace layout -------
  char* ws = (char*)d_ws;
  size_t off = 0;
  auto alloc = [&](size_t bytes) -> char* {
    char* p = ws + off;
    off = (off + bytes + 255) & ~(size_t)255;
    return p;
  };
  float* sums   = (float*)alloc((size_t)2 * BATCH * GRID3 * sizeof(float));
  float* cnts   = sums + (size_t)BATCH * GRID3;
  float* feat   = (float*)alloc((size_t)BATCH * GRID3 * sizeof(float));
  float* occ64  = (float*)alloc((size_t)BATCH * GRID3 * sizeof(float));
  float* occ32  = (float*)alloc((size_t)BATCH * 32 * 32 * 32 * sizeof(float));
  float* occ16  = (float*)alloc((size_t)BATCH * 16 * 16 * 16 * sizeof(float));
  float* occ8   = (float*)alloc((size_t)BATCH * 8 * 8 * 8 * sizeof(float));
  float* occ4   = (float*)alloc((size_t)BATCH * 4 * 4 * 4 * sizeof(float));
  float* stats  = (float*)alloc((2 * 512 + 1) * sizeof(float));
  float* actF32 = (float*)alloc((size_t)BATCH * GRID3 * 32 * sizeof(float));
  _Float16* actF16 = (_Float16*)alloc((size_t)BATCH * GRID3 * 32 * sizeof(_Float16));
  _Float16* wpk[11];
  wpk[0] = nullptr;
  for (int i = 1; i < 11; ++i) {
    int taps = (ks[i] == 3) ? 27 : 1;
    size_t elems = (size_t)taps * cins[i] * couts[i];
    wpk[i] = (_Float16*)alloc(elems * sizeof(_Float16));
  }

  auto occAt = [&](int S) -> float* {
    switch (S) {
      case 64: return occ64;
      case 32: return occ32;
      case 16: return occ16;
      case 8:  return occ8;
      default: return occ4;
    }
  };
  auto ilog2 = [](int v) { int l = 0; while ((1 << l) < v) ++l; return l; };

  // ------- voxelize -------
  {
    int n = 2 * BATCH * GRID3;
    zero_kernel<<<(n + 255) / 256, 256, 0, stream>>>(sums, n);
    int np = BATCH * Npts;
    vox_scatter_kernel<<<(np + 255) / 256, 256, 0, stream>>>(points, sums, cnts, Npts);
    int nv = BATCH * GRID3;
    vox_finalize_kernel<<<(nv + 255) / 256, 256, 0, stream>>>(sums, cnts, feat, occ64);
  }

  // ------- pack WMMA weights for layers 1..10 -------
  for (int i = 1; i < 11; ++i) {
    int taps = (ks[i] == 3) ? 27 : 1;
    long long total = (long long)taps * cins[i] * couts[i];
    pack_weights_kernel<<<(int)((total + 255) / 256), 256, 0, stream>>>(
        W[i], wpk[i], cins[i], couts[i], taps);
  }

  // ------- layer 0 conv (cin=1, direct) -------
  {
    int nv = BATCH * GRID3;
    conv0_kernel<<<(nv + 255) / 256, 256, 0, stream>>>(feat, W[0], actF32);
  }

  // ------- per-layer: (occ downsample) + BN stats/apply + next conv -------
  for (int i = 0; i < 11; ++i) {
    int So = Sout[i], C = couts[i];
    int logC = ilog2(C);
    int nvox = BATCH * So * So * So;

    if (ss[i] == 2) {
      occ_down_kernel<<<(nvox + 255) / 256, 256, 0, stream>>>(
          occAt(Sin[i]), occAt(So), ilog2(So));
    }
    float* occ = occAt(So);

    int nst = 2 * C + 1;
    zero_kernel<<<(nst + 255) / 256, 256, 0, stream>>>(stats, nst);
    long long total = (long long)nvox * C;
    int sblocks = (int)((total + 255) / 256);
    if (sblocks > 512) sblocks = 512;
    bn_stats_kernel<<<sblocks, 256, 0, stream>>>(actF32, occ, stats, nvox, C, logC);
    bn_apply_kernel<<<(int)((total + 255) / 256), 256, 0, stream>>>(
        actF32, occ, stats, Gm[i], Bt[i], actF16, nvox, C, logC);

    if (i < 10) {
      int j = i + 1;
      int taps = (ks[j] == 3) ? 27 : 1;
      int SoN = Sout[j];
      int mtiles = (BATCH * SoN * SoN * SoN) / 16;
      if (couts[j] == 32) {
        dim3 grid(mtiles, couts[j] / 32);   // NPW = 2
        conv_wmma_kernel<2><<<grid, 32, 0, stream>>>(
            actF16, wpk[j], actF32, ilog2(Sin[j]), ilog2(SoN), ss[j], taps,
            cins[j], couts[j]);
      } else {
        dim3 grid(mtiles, couts[j] / 64);   // NPW = 4
        conv_wmma_kernel<4><<<grid, 32, 0, stream>>>(
            actF16, wpk[j], actF32, ilog2(Sin[j]), ilog2(SoN), ss[j], taps,
            cins[j], couts[j]);
      }
    }
  }

  // ------- final masked global average pool -------
  pool_kernel<<<BATCH, 512, 0, stream>>>(actF16, occ4, (float*)d_out);
}
